// TransformerDecoder_34402688041089
// MI455X (gfx1250) — matrix operand order repaired and checked by reading, not compiled, not run on hardware
//
#include <hip/hip_runtime.h>
#include <cstdint>
#include <cmath>

// ---------------- geometry ----------------
static constexpr int BB   = 2;      // batch
static constexpr int NQ   = 200;    // queries
static constexpr int QP   = 201;    // queries + presence token
static constexpr int QPP  = 224;    // QP padded to multiple of 32 (self-attn K dim)
static constexpr int DM   = 256;    // d_model
static constexpr int NH   = 8;      // heads
static constexpr int HD   = 32;     // head dim
static constexpr int FF   = 2048;   // ffn dim
static constexpr int HSZ  = 72;
static constexpr int WSZ  = 72;
static constexpr int LL   = HSZ * WSZ;    // 5184 memory tokens
static constexpr float INV_SQRT_HD = 0.17677669529663687f;

static constexpr int LAY_ROW   = 0;   // C[z*sC + r*ldc + c]
static constexpr int LAY_HS    = 1;   // head-split  [b][h][t][d]
static constexpr int LAY_HST   = 2;   // head-split transposed [b][h][d][t]
static constexpr int LAY_MERGE = 3;   // merge heads [b][t][h*32+d], z=(b,h)

typedef __attribute__((ext_vector_type(8)))  _Float16 v8h;
typedef __attribute__((ext_vector_type(16))) _Float16 v16h;
typedef __attribute__((ext_vector_type(8)))  float    v8f;

static __device__ __forceinline__ v16h cat16(v8h lo, v8h hi) {
  return __builtin_shufflevector(lo, hi, 0,1,2,3,4,5,6,7,8,9,10,11,12,13,14,15);
}

// ---------------- elementwise helpers ----------------
__global__ void k_cvt(_Float16* __restrict__ dst, const float* __restrict__ a,
                      const float* __restrict__ b, long n) {
  long i = (long)blockIdx.x * blockDim.x + threadIdx.x;
  if (i >= n) return;
  float v = a[i];
  if (b) v += b[i];
  dst[i] = (_Float16)v;
}

__global__ void k_sigref(float* __restrict__ refp, const float* __restrict__ rp) {
  int i = blockIdx.x * blockDim.x + threadIdx.x;
  if (i >= BB * NQ * 4) return;
  int j = i % (NQ * 4);
  refp[i] = 1.f / (1.f + expf(-rp[j]));
}

__global__ void k_init_tgt(float* __restrict__ tgt, const float* __restrict__ pres,
                           const float* __restrict__ qemb) {
  int i = blockIdx.x * blockDim.x + threadIdx.x;
  if (i >= BB * QP * DM) return;
  int d = i & 255;
  int t = (i >> 8) % QP;
  tgt[i] = (t == 0) ? pres[d] : qemb[(t - 1) * DM + d];
}

__global__ void k_build_pos(float* __restrict__ posp, const float* __restrict__ qpos) {
  int i = blockIdx.x * blockDim.x + threadIdx.x;
  if (i >= BB * QP * DM) return;
  int d = i & 255;
  int t = (i >> 8) % QP;
  int b = i / (QP * DM);
  posp[i] = (t == 0) ? 0.f : qpos[((long)(b * NQ + t - 1)) * DM + d];
}

__global__ void k_sine(float* __restrict__ out, const float* __restrict__ refp) {
  int i = blockIdx.x * blockDim.x + threadIdx.x;
  if (i >= BB * NQ * 4 * 128) return;
  int f  = i & 127;
  int pd = (i >> 7) & 3;
  int qq = i >> 9;                       // b*NQ + q
  float pos  = refp[qq * 4 + pd];
  float e    = (float)(2 * (f >> 1)) * (1.f / 128.f);
  float freq = powf(10000.f, e);
  float raw  = pos * (2.f * 3.14159265358979323846f) / freq;
  out[i] = (f & 1) ? cosf(raw) : sinf(raw);
}

__global__ void k_refupd(float* __restrict__ dst, const float* __restrict__ refp,
                         const float* __restrict__ delta, int n) {
  int i = blockIdx.x * blockDim.x + threadIdx.x;
  if (i >= n) return;
  float r = refp[i];
  float x = logf(r / (1.f - r + 1e-6f) + 1e-6f) + delta[i];
  dst[i] = 1.f / (1.f + expf(-x));
}

// ---------------- generic fp16 WMMA GEMM: C = A(MxK) * W(NxK)^T ----------------
// Out-of-range rows are CLAMPED (not predicated): an OOB A-row only feeds C-rows
// that are never stored, and an OOB B-row only feeds the C-column owned by that
// lane, which is also never stored. This keeps the K-loop free of any EXEC
// manipulation: pure load->wmma.
__global__ __launch_bounds__(128)
void k_gemm(const _Float16* __restrict__ A, int lda, long sA,
            const _Float16* __restrict__ Wm, int ldw, long sW,
            const float* __restrict__ bias, void* __restrict__ Cv,
            int M, int N, int K,
            int layout, int ldc, long sC, int T, int ldT,
            int relu, int outHalf, float scale) {
  int z = blockIdx.z;
  const _Float16* Ab = A + (long)z * sA;
  const _Float16* Wb = Wm + (long)z * sW;
  int wv   = threadIdx.x >> 5;
  int lane = threadIdx.x & 31;
  int tm = blockIdx.x * 16;
  int tn = (blockIdx.y * 4 + wv) * 16;
  if (tn >= N) return;
  int hi  = lane >> 4;
  int l15 = lane & 15;
  int ar = tm + l15; if (ar >= M) ar = M - 1;   // clamp (see note above)
  int br = tn + l15; if (br >= N) br = N - 1;
  const _Float16* Ap = Ab + (long)ar * lda + hi * 8;    // A: chunks at k0, k0+16
  const _Float16* Wp = Wb + (long)br * ldw + hi * 8;    // B: chunks at k0, k0+16
  v8f acc = {};
  for (int k0 = 0; k0 < K; k0 += 32) {
    v8h a0 = *reinterpret_cast<const v8h*>(Ap + k0);
    v8h a1 = *reinterpret_cast<const v8h*>(Ap + k0 + 16);
    v8h b0 = *reinterpret_cast<const v8h*>(Wp + k0);
    v8h b1 = *reinterpret_cast<const v8h*>(Wp + k0 + 16);
    __builtin_prefetch(Ap + k0 + 128, 0, 1);   // speculative; OOB is dropped
    __builtin_prefetch(Wp + k0 + 128, 0, 1);
    acc = __builtin_amdgcn_wmma_f32_16x16x32_f16(false, cat16(a0, a1),
                                                 false, cat16(b0, b1),
                                                 (short)0, acc, false, false);
  }
  float*    Cf = (float*)Cv;
  _Float16* Ch = (_Float16*)Cv;
  int cn = tn + l15;
#pragma unroll
  for (int i = 0; i < 8; ++i) {
    int r = tm + i + hi * 8;
    if (r >= M || cn >= N) continue;
    float v = acc[i] * scale;
    if (bias) v += bias[cn];
    if (relu) v = fmaxf(v, 0.f);
    long off;
    if (layout == LAY_ROW) {
      off = (long)z * sC + (long)r * ldc + cn;
    } else if (layout == LAY_HS) {
      int bb = r / T, t = r % T, h = cn >> 5, d = cn & 31;
      off = (((long)(bb * NH + h) * T) + t) * HD + d;
    } else if (layout == LAY_HST) {
      int bb = r / T, t = r % T, h = cn >> 5, d = cn & 31;
      off = ((long)(bb * NH + h) * HD + d) * (long)ldT + t;
    } else { // LAY_MERGE: N==32, z=(b*NH+h)
      int bb = z / NH, h = z % NH;
      off = ((long)(bb * T + r)) * ldc + h * HD + cn;
    }
    if (outHalf) Ch[off] = (_Float16)v; else Cf[off] = v;
  }
}

// -------- cross-attention scores with fused box-RPB bias (K=32, one WMMA) --------
__global__ __launch_bounds__(128)
void k_score_cross(const _Float16* __restrict__ Q, const _Float16* __restrict__ Km,
                   float* __restrict__ S,
                   const float* __restrict__ rpbx, const float* __restrict__ rpby) {
  int z = blockIdx.z;                 // b*NH + h
  int b = z / NH, h = z % NH;
  const _Float16* Qb = Q  + (long)z * QP * HD;
  const _Float16* Kb = Km + (long)z * LL * HD;
  int wv   = threadIdx.x >> 5;
  int lane = threadIdx.x & 31;
  int tm = blockIdx.x * 16;
  int tn = (blockIdx.y * 4 + wv) * 16;   // LL % 64 == 0
  int hi  = lane >> 4;
  int l15 = lane & 15;
  int ar = tm + l15; if (ar >= QP) ar = QP - 1;  // clamp; OOB rows never stored
  int br = tn + l15;
  const _Float16* Ap = Qb + (long)ar * HD + hi * 8;
  v8h a0 = *reinterpret_cast<const v8h*>(Ap);
  v8h a1 = *reinterpret_cast<const v8h*>(Ap + 16);
  const _Float16* Kp = Kb + (long)br * HD + hi * 8;
  v8h b0 = *reinterpret_cast<const v8h*>(Kp);
  v8h b1 = *reinterpret_cast<const v8h*>(Kp + 16);
  v8f acc = {};
  acc = __builtin_amdgcn_wmma_f32_16x16x32_f16(false, cat16(a0, a1),
                                               false, cat16(b0, b1),
                                               (short)0, acc, false, false);
  int cn = tn + l15;
#pragma unroll
  for (int i = 0; i < 8; ++i) {
    int r = tm + i + hi * 8;
    if (r >= QP) continue;
    float v = acc[i] * INV_SQRT_HD;
    if (r > 0) {                        // presence row gets zero bias
      int q  = r - 1;
      int ky = cn / WSZ, kx = cn % WSZ;
      v += rpby[(((long)(b * NQ + q)) * HSZ + ky) * NH + h]
         + rpbx[(((long)(b * NQ + q)) * WSZ + kx) * NH + h];
    }
    S[((long)z * QP + r) * LL + cn] = v;
  }
}

// ---------------- softmax row -> fp16 probabilities (zero-pads) ----------------
__global__ __launch_bounds__(256)
void k_softmax(const float* __restrict__ S, int srow,
               _Float16* __restrict__ P, int prow, int ncol, int npad) {
  long row = blockIdx.x;
  const float* s = S + row * (long)srow;
  _Float16*    p = P + row * (long)prow;
  __shared__ float red[256];
  int t = threadIdx.x;
  float m = -3.4e38f;
  for (int c = t; c < ncol; c += 256) m = fmaxf(m, s[c]);
  red[t] = m; __syncthreads();
  for (int st = 128; st > 0; st >>= 1) { if (t < st) red[t] = fmaxf(red[t], red[t + st]); __syncthreads(); }
  m = red[0]; __syncthreads();
  float sum = 0.f;
  for (int c = t; c < ncol; c += 256) sum += expf(s[c] - m);
  red[t] = sum; __syncthreads();
  for (int st = 128; st > 0; st >>= 1) { if (t < st) red[t] += red[t + st]; __syncthreads(); }
  float inv = 1.f / red[0];
  for (int c = t; c < npad; c += 256)
    p[c] = (c < ncol) ? (_Float16)(expf(s[c] - m) * inv) : (_Float16)0.f;
}

// ---------------- layernorm (D=256), dst = LN(a (+ b)) * g + be ----------------
__global__ __launch_bounds__(256)
void k_ln(float* __restrict__ dst, int ldd,
          const float* __restrict__ a, int lda,
          const float* __restrict__ b, int ldb,
          const float* __restrict__ g, const float* __restrict__ be) {
  int row = blockIdx.x;
  int t = threadIdx.x;
  float v = a[(long)row * lda + t];
  if (b) v += b[(long)row * ldb + t];
  __shared__ float red[256];
  red[t] = v; __syncthreads();
  for (int st = 128; st > 0; st >>= 1) { if (t < st) red[t] += red[t + st]; __syncthreads(); }
  float mean = red[0] * (1.f / 256.f); __syncthreads();
  float d = v - mean;
  red[t] = d * d; __syncthreads();
  for (int st = 128; st > 0; st >>= 1) { if (t < st) red[t] += red[t + st]; __syncthreads(); }
  float var = red[0] * (1.f / 256.f);
  dst[(long)row * ldd + t] = d * rsqrtf(var + 1e-5f) * g[t] + be[t];
}

// ---------------- fused box-RPB MLP (2 -> 256 relu -> 8), one wave/position ----
static __device__ __forceinline__ float logscale(float d) {
  float v = d * 8.f;
  float s = (v > 0.f) ? 1.f : ((v < 0.f) ? -1.f : 0.f);
  return s * log2f(fabsf(v) + 1.f) * (1.f / 3.f);   // /log2(8)
}

__global__ __launch_bounds__(256)
void k_rpb(float* __restrict__ out, const float* __restrict__ refp,
           const float* __restrict__ w0, const float* __restrict__ b0,
           const float* __restrict__ w1, const float* __restrict__ b1, int axis) {
  int wv   = threadIdx.x >> 5;
  int lane = threadIdx.x & 31;
  int gid = blockIdx.x * 8 + wv;          // over B*NQ*72 positions
  if (gid >= BB * NQ * 72) return;
  int k  = gid % 72;
  int bq = gid / 72;
  const float* r = refp + bq * 4;
  float c  = (axis == 0) ? r[0] : r[1];
  float sz = (axis == 0) ? r[2] : r[3];
  float coord = (float)k * (1.f / 72.f);
  float l0 = logscale(coord - (c - 0.5f * sz));
  float l1 = logscale(coord - (c + 0.5f * sz));
  float hbuf[8];
#pragma unroll
  for (int u = 0; u < 8; ++u) {
    int n = lane * 8 + u;
    hbuf[u] = fmaxf(w0[n * 2 + 0] * l0 + w0[n * 2 + 1] * l1 + b0[n], 0.f);
  }
#pragma unroll
  for (int j = 0; j < 8; ++j) {
    float p = 0.f;
#pragma unroll
    for (int u = 0; u < 8; ++u) p += w1[j * 256 + lane * 8 + u] * hbuf[u];
    for (int m = 16; m >= 1; m >>= 1) p += __shfl_xor(p, m, 32);
    if (lane == 0) out[(long)gid * 8 + j] = p + b1[j];
  }
}

// ---------------- tiny output heads (N=4 / VALU) ----------------
__global__ void k_head4(float* __restrict__ out, const float* __restrict__ hin,
                        const float* __restrict__ w, const float* __restrict__ bias,
                        int rows, int nout) {
  int i = blockIdx.x * blockDim.x + threadIdx.x;
  if (i >= rows * nout) return;
  int r = i / nout, j = i % nout;
  const float* h  = hin + (long)r * DM;
  const float* wr = w + (long)j * DM;
  float s = bias[j];
  for (int n = 0; n < DM; ++n) s += wr[n] * h[n];
  out[i] = s;
}

// ---------------- presence head: LN + MLP(256->256->256->1) ----------------
__global__ __launch_bounds__(256)
void k_presence(float* __restrict__ out, const float* __restrict__ x, int xstride,
                const float* __restrict__ g, const float* __restrict__ be,
                const float* __restrict__ w0, const float* __restrict__ b0,
                const float* __restrict__ w1, const float* __restrict__ b1,
                const float* __restrict__ w2, const float* __restrict__ b2) {
  int b = blockIdx.x, t = threadIdx.x;
  __shared__ float xs[256], hs[256], red[256];
  float v = x[(long)b * xstride + t];
  red[t] = v; __syncthreads();
  for (int st = 128; st > 0; st >>= 1) { if (t < st) red[t] += red[t + st]; __syncthreads(); }
  float mean = red[0] * (1.f / 256.f); __syncthreads();
  float d = v - mean;
  red[t] = d * d; __syncthreads();
  for (int st = 128; st > 0; st >>= 1) { if (t < st) red[t] += red[t + st]; __syncthreads(); }
  float var = red[0] * (1.f / 256.f); __syncthreads();
  xs[t] = d * rsqrtf(var + 1e-5f) * g[t] + be[t];
  __syncthreads();
  float h1 = b0[t];
  for (int n = 0; n < 256; ++n) h1 += w0[t * 256 + n] * xs[n];
  hs[t] = fmaxf(h1, 0.f); __syncthreads();
  float h2 = b1[t];
  for (int n = 0; n < 256; ++n) h2 += w1[t * 256 + n] * hs[n];
  __syncthreads();
  xs[t] = fmaxf(h2, 0.f); __syncthreads();
  red[t] = w2[t] * xs[t]; __syncthreads();
  for (int st = 128; st > 0; st >>= 1) { if (t < st) red[t] += red[t + st]; __syncthreads(); }
  if (t == 0) out[b] = red[0] + b2[0];
}

// ================= host orchestration =================
struct LinW { const float* w; const float* b; };
struct MHAW { LinW q, k, v, o; };
struct LayerW { MHAW sa, ca; LinW n1, n2, n3, l1, l2; };
struct LayerH { _Float16 *saq,*sak,*sav,*sao,*caq,*cak,*cav,*cao,*l1,*l2; };

extern "C" void kernel_launch(void* const* d_in, const int* in_sizes, int n_in,
                              void* d_out, int out_size, void* d_ws, size_t ws_size,
                              hipStream_t stream) {
  (void)in_sizes; (void)n_in; (void)out_size; (void)ws_size;
  int ip = 0;
  auto nxt = [&]() -> const float* { return (const float*)d_in[ip++]; };
  auto lin = [&]() -> LinW { LinW l; l.w = nxt(); l.b = nxt(); return l; };
  auto mha = [&]() -> MHAW { MHAW m; m.q = lin(); m.k = lin(); m.v = lin(); m.o = lin(); return m; };

  const float* memory     = nxt();
  const float* memory_pos = nxt();
  LayerW Ls[6];
  for (int li = 0; li < 6; ++li) {
    Ls[li].sa = mha(); Ls[li].ca = mha();
    Ls[li].n1 = lin(); Ls[li].n2 = lin(); Ls[li].n3 = lin();
    Ls[li].l1 = lin(); Ls[li].l2 = lin();
  }
  LinW normW = lin();
  const float* query_embed      = nxt();
  const float* reference_points = nxt();
  LinW rph0 = lin(), rph1 = lin();
  LinW bb0 = lin(), bb1 = lin(), bb2 = lin();
  LinW rpx0 = lin(), rpx1 = lin();
  LinW rpy0 = lin(), rpy1 = lin();
  const float* presence_token = nxt();
  LinW ph0 = lin(), ph1 = lin(), ph2 = lin();
  LinW pnorm = lin();
  // trailing H, W scalars ignored (compile-time 72)

  // ---------------- workspace carve ----------------
  uint8_t* base = (uint8_t*)d_ws;
  size_t off = 0;
  auto alloc = [&](size_t bytes) -> void* {
    void* p = base + off;
    off += (bytes + 255) & ~(size_t)255;
    return p;
  };
  const long NMEM = (long)BB * LL * DM;          // 2,654,208
  const long NX   = (long)BB * QP * DM;          // 102,912
  const long NQD  = (long)BB * NQ * DM;          // 102,400
  const long NP   = (long)BB * NH * QP * LL;     // 16,671,744

  _Float16* h_mem  = (_Float16*)alloc(NMEM * 2);
  _Float16* h_memk = (_Float16*)alloc(NMEM * 2);
  _Float16* h_actA = (_Float16*)alloc((size_t)1048576 * 2);
  _Float16* h_actB = (_Float16*)alloc((size_t)1048576 * 2);
  _Float16* h_P    = (_Float16*)alloc(NP * 2);
  float* S      = (float*)alloc(NP * 4);
  float* tgt    = (float*)alloc(NX * 4);
  float* posp   = (float*)alloc(NX * 4);
  float* attn_o = (float*)alloc(NX * 4);
  float* qpos   = (float*)alloc(NQD * 4);
  float* sine   = (float*)alloc((size_t)BB * NQ * 512 * 4);
  float* tmp0   = (float*)alloc(NQD * 4);
  float* tmp1   = (float*)alloc(NQD * 4);
  float* lnq    = (float*)alloc(NQD * 4);
  float* ffn_h  = (float*)alloc((size_t)BB * QP * FF * 4);
  float* refp   = (float*)alloc((size_t)BB * NQ * 4 * 4);
  float* delta  = (float*)alloc((size_t)BB * NQ * 4 * 4);
  float* rpbx   = (float*)alloc((size_t)BB * NQ * WSZ * NH * 4);
  float* rpby   = (float*)alloc((size_t)BB * NQ * HSZ * NH * 4);
  _Float16* qs  = (_Float16*)alloc((size_t)BB * NH * QP * HD * 2);
  _Float16* ks  = (_Float16*)alloc((size_t)BB * NH * QP * HD * 2);
  _Float16* vts = (_Float16*)alloc((size_t)BB * NH * HD * QPP * 2);
  _Float16* qc  = (_Float16*)alloc((size_t)BB * NH * QP * HD * 2);
  _Float16* kc  = (_Float16*)alloc((size_t)BB * NH * LL * HD * 2);
  _Float16* vtc = (_Float16*)alloc((size_t)BB * NH * HD * LL * 2);
  _Float16* hw  = (_Float16*)alloc((size_t)9764864 * 2);

  auto cvt = [&](_Float16* dst, const float* a, const float* b, long n) {
    k_cvt<<<(int)((n + 255) / 256), 256, 0, stream>>>(dst, a, b, n);
  };
  size_t wo = 0;
  auto carve = [&](size_t n, const float* src) -> _Float16* {
    _Float16* p = hw + wo; wo += n; cvt(p, src, nullptr, (long)n); return p;
  };
  LayerH lh[6];
  for (int li = 0; li < 6; ++li) {
    lh[li].saq = carve(65536, Ls[li].sa.q.w);
    lh[li].sak = carve(65536, Ls[li].sa.k.w);
    lh[li].sav = carve(65536, Ls[li].sa.v.w);
    lh[li].sao = carve(65536, Ls[li].sa.o.w);
    lh[li].caq = carve(65536, Ls[li].ca.q.w);
    lh[li].cak = carve(65536, Ls[li].ca.k.w);
    lh[li].cav = carve(65536, Ls[li].ca.v.w);
    lh[li].cao = carve(65536, Ls[li].ca.o.w);
    lh[li].l1  = carve((size_t)FF * DM, Ls[li].l1.w);
    lh[li].l2  = carve((size_t)DM * FF, Ls[li].l2.w);
  }
  _Float16* h_rph0 = carve(256 * 512, rph0.w);
  _Float16* h_rph1 = carve(65536, rph1.w);
  _Float16* h_bb0  = carve(65536, bb0.w);
  _Float16* h_bb1  = carve(65536, bb1.w);

  auto gemm = [&](const _Float16* A, int lda, long sA,
                  const _Float16* Wm, int ldw, long sW,
                  const float* bias, void* C,
                  int M, int N, int K, int layout, int ldc, long sC,
                  int T, int ldT, int relu, int outHalf, float scale, int nz) {
    dim3 g((M + 15) / 16, (N + 63) / 64, nz);
    k_gemm<<<g, 128, 0, stream>>>(A, lda, sA, Wm, ldw, sW, bias, C, M, N, K,
                                  layout, ldc, sC, T, ldT, relu, outHalf, scale);
  };

  // ---------------- prologue ----------------
  k_sigref<<<(BB * NQ * 4 + 255) / 256, 256, 0, stream>>>(refp, reference_points);
  k_init_tgt<<<(int)((NX + 255) / 256), 256, 0, stream>>>(tgt, presence_token, query_embed);
  cvt(h_mem,  memory, nullptr, NMEM);
  cvt(h_memk, memory, memory_pos, NMEM);

  // ---------------- decoder layers ----------------
  for (int li = 0; li < 6; ++li) {
    const LayerW& L = Ls[li];
    const LayerH& Hh = lh[li];
    // query_pos = MLP(sine_embed(ref))
    k_sine<<<(BB * NQ * 512 + 255) / 256, 256, 0, stream>>>(sine, refp);
    cvt(h_actA, sine, nullptr, (long)BB * NQ * 512);
    gemm(h_actA, 512, 0, h_rph0, 512, 0, rph0.b, tmp0, BB * NQ, DM, 512,
         LAY_ROW, DM, 0, 0, 0, 1, 0, 1.f, 1);
    cvt(h_actA, tmp0, nullptr, NQD);
    gemm(h_actA, DM, 0, h_rph1, DM, 0, rph1.b, qpos, BB * NQ, DM, DM,
         LAY_ROW, DM, 0, 0, 0, 0, 0, 1.f, 1);
    k_build_pos<<<(int)((NX + 255) / 256), 256, 0, stream>>>(posp, qpos);
    // box RPB tables
    k_rpb<<<(BB * NQ * 72 + 7) / 8, 256, 0, stream>>>(rpbx, refp, rpx0.w, rpx0.b, rpx1.w, rpx1.b, 0);
    k_rpb<<<(BB * NQ * 72 + 7) / 8, 256, 0, stream>>>(rpby, refp, rpy0.w, rpy0.b, rpy1.w, rpy1.b, 1);
    // ---- self attention ----
    cvt(h_actA, tgt, posp, NX);            // q/k input = x + pos
    cvt(h_actB, tgt, nullptr, NX);         // v input = x
    gemm(h_actA, DM, 0, Hh.saq, DM, 0, L.sa.q.b, qs, BB * QP, DM, DM, LAY_HS, 0, 0, QP, 0, 0, 1, 1.f, 1);
    gemm(h_actA, DM, 0, Hh.sak, DM, 0, L.sa.k.b, ks, BB * QP, DM, DM, LAY_HS, 0, 0, QP, 0, 0, 1, 1.f, 1);
    gemm(h_actB, DM, 0, Hh.sav, DM, 0, L.sa.v.b, vts, BB * QP, DM, DM, LAY_HST, 0, 0, QP, QPP, 0, 1, 1.f, 1);
    gemm(qs, HD, (long)QP * HD, ks, HD, (long)QP * HD, nullptr, S, QP, QP, HD,
         LAY_ROW, QPP, (long)QP * QPP, 0, 0, 0, 0, INV_SQRT_HD, BB * NH);
    k_softmax<<<BB * NH * QP, 256, 0, stream>>>(S, QPP, h_P, QPP, QP, QPP);
    gemm(h_P, QPP, (long)QP * QPP, vts, QPP, (long)HD * QPP, nullptr, h_actB,
         QP, HD, QPP, LAY_MERGE, DM, 0, QP, 0, 0, 1, 1.f, BB * NH);
    gemm(h_actB, DM, 0, Hh.sao, DM, 0, L.sa.o.b, attn_o, BB * QP, DM, DM, LAY_ROW, DM, 0, 0, 0, 0, 0, 1.f, 1);
    k_ln<<<BB * QP, 256, 0, stream>>>(tgt, DM, tgt, DM, attn_o, DM, L.n2.w, L.n2.b);
    // ---- cross attention ----
    cvt(h_actA, tgt, posp, NX);
    gemm(h_actA, DM, 0, Hh.caq, DM, 0, L.ca.q.b, qc, BB * QP, DM, DM, LAY_HS, 0, 0, QP, 0, 0, 1, 1.f, 1);
    gemm(h_memk, DM, 0, Hh.cak, DM, 0, L.ca.k.b, kc, BB * LL, DM, DM, LAY_HS, 0, 0, LL, 0, 0, 1, 1.f, 1);
    gemm(h_mem,  DM, 0, Hh.cav, DM, 0, L.ca.v.b, vtc, BB * LL, DM, DM, LAY_HST, 0, 0, LL, LL, 0, 1, 1.f, 1);
    {
      dim3 g((QP + 15) / 16, LL / 64, BB * NH);
      k_score_cross<<<g, 128, 0, stream>>>(qc, kc, S, rpbx, rpby);
    }
    k_softmax<<<BB * NH * QP, 256, 0, stream>>>(S, LL, h_P, LL, LL, LL);
    gemm(h_P, LL, (long)QP * LL, vtc, LL, (long)HD * LL, nullptr, h_actB,
         QP, HD, LL, LAY_MERGE, DM, 0, QP, 0, 0, 1, 1.f, BB * NH);
    gemm(h_actB, DM, 0, Hh.cao, DM, 0, L.ca.o.b, attn_o, BB * QP, DM, DM, LAY_ROW, DM, 0, 0, 0, 0, 0, 1.f, 1);
    k_ln<<<BB * QP, 256, 0, stream>>>(tgt, DM, tgt, DM, attn_o, DM, L.n1.w, L.n1.b);
    // ---- FFN ----
    cvt(h_actA, tgt, nullptr, NX);
    gemm(h_actA, DM, 0, Hh.l1, DM, 0, L.l1.b, ffn_h, BB * QP, FF, DM, LAY_ROW, FF, 0, 0, 0, 1, 0, 1.f, 1);
    cvt(h_actB, ffn_h, nullptr, (long)BB * QP * FF);
    gemm(h_actB, FF, 0, Hh.l2, FF, 0, L.l2.b, attn_o, BB * QP, DM, FF, LAY_ROW, DM, 0, 0, 0, 0, 0, 1.f, 1);
    k_ln<<<BB * QP, 256, 0, stream>>>(tgt, DM, tgt, DM, attn_o, DM, L.n3.w, L.n3.b);
    // ---- reference point update ----
    if (li < 5) {
      for (int b = 0; b < BB; ++b)
        k_ln<<<NQ, 256, 0, stream>>>(lnq + (long)b * NQ * DM, DM,
                                     tgt + ((long)b * QP + 1) * DM, DM,
                                     nullptr, 0, normW.w, normW.b);
      cvt(h_actA, lnq, nullptr, NQD);
      gemm(h_actA, DM, 0, h_bb0, DM, 0, bb0.b, tmp0, BB * NQ, DM, DM, LAY_ROW, DM, 0, 0, 0, 1, 0, 1.f, 1);
      cvt(h_actA, tmp0, nullptr, NQD);
      gemm(h_actA, DM, 0, h_bb1, DM, 0, bb1.b, tmp1, BB * NQ, DM, DM, LAY_ROW, DM, 0, 0, 0, 1, 0, 1.f, 1);
      k_head4<<<(BB * NQ * 4 + 255) / 256, 256, 0, stream>>>(delta, tmp1, bb2.w, bb2.b, BB * NQ, 4);
      k_refupd<<<(BB * NQ * 4 + 255) / 256, 256, 0, stream>>>(refp, refp, delta, BB * NQ * 4);
    }
  }

  // ---------------- outputs ----------------
  float* outq    = (float*)d_out;
  float* outbox  = outq + (long)BB * NQ * DM;
  float* outpres = outbox + (long)BB * NQ * 4;
  for (int b = 0; b < BB; ++b)
    k_ln<<<NQ, 256, 0, stream>>>(outq + (long)b * NQ * DM, DM,
                                 tgt + ((long)b * QP + 1) * DM, DM,
                                 nullptr, 0, normW.w, normW.b);
  cvt(h_actA, outq, nullptr, NQD);
  gemm(h_actA, DM, 0, h_bb0, DM, 0, bb0.b, tmp0, BB * NQ, DM, DM, LAY_ROW, DM, 0, 0, 0, 1, 0, 1.f, 1);
  cvt(h_actA, tmp0, nullptr, NQD);
  gemm(h_actA, DM, 0, h_bb1, DM, 0, bb1.b, tmp1, BB * NQ, DM, DM, LAY_ROW, DM, 0, 0, 0, 1, 0, 1.f, 1);
  k_head4<<<(BB * NQ * 4 + 255) / 256, 256, 0, stream>>>(delta, tmp1, bb2.w, bb2.b, BB * NQ, 4);
  k_refupd<<<(BB * NQ * 4 + 255) / 256, 256, 0, stream>>>(outbox, refp, delta, BB * NQ * 4);
  k_presence<<<BB, 256, 0, stream>>>(outpres, tgt, QP * DM, pnorm.w, pnorm.b,
                                     ph0.w, ph0.b, ph1.w, ph1.b, ph2.w, ph2.b);
}
